// Head_10548439679510
// MI455X (gfx1250) — matrix operand order repaired and checked by reading, not compiled, not run on hardware
//
#include <hip/hip_runtime.h>
#include <stdint.h>

typedef __attribute__((ext_vector_type(16))) __bf16 v16bf;
typedef __attribute__((ext_vector_type(8)))  float  v8f;

#define NB 8
#define NT 2048
#define NE 1024
#define NH 64
#define SOFT_SCALE 0.03125f   // 1/sqrt(E) = 1/32 (reference scales by n_embd**-0.5)

#if defined(__has_builtin)
#if __has_builtin(__builtin_amdgcn_tensor_load_to_lds) && \
    __has_builtin(__builtin_amdgcn_s_wait_tensorcnt)
#define USE_TDM 1
#endif
#endif
#ifndef USE_TDM
#define USE_TDM 0
#endif

// ---------- bf16 helpers ----------
static __device__ __forceinline__ unsigned short f2bf(float f) {
  union { float f; uint32_t u; } x; x.f = f;
  uint32_t r = x.u + 0x7FFFu + ((x.u >> 16) & 1u);   // round-to-nearest-even
  return (unsigned short)(r >> 16);
}
static __device__ __forceinline__ uint32_t pack_bf2(float lo, float hi) {
  return (uint32_t)f2bf(lo) | ((uint32_t)f2bf(hi) << 16);
}

union FragBF { v16bf v; uint32_t u32[8]; };

static __device__ __forceinline__ v8f bf16_wmma(v16bf a, v16bf b, v8f c) {
  return __builtin_amdgcn_wmma_f32_16x16x32_bf16(false, a, false, b, (short)0, c,
                                                 false, false);
}

// ---------- WMMA fragment loaders (contiguous k = 16*half + s in A and B, wave32) ----------
// A fragment: 16x32 bf16, rows row0.., k-offset k0, row-major u16 src, ld elems.
static __device__ __forceinline__ v16bf load_frag_a(const unsigned short* s, int ld,
                                                    int row0, int k0) {
  int lane = (int)(threadIdx.x & 31);
  int half = lane >> 4;
  int m = row0 + (lane & 15);
  FragBF f;
#pragma unroll
  for (int j = 0; j < 8; ++j) {
    int kb = half * 16 + 2 * j;
    f.u32[j] = *(const uint32_t*)(s + (size_t)m * ld + k0 + kb);
  }
  return f.v;
}

// B fragment 32x16 from a ROW-PAIR-PACKED u32 source:
//   s[kp*ld + n] = pack(B[2kp][n], B[2kp+1][n]).  k0 in elements (even).
static __device__ __forceinline__ v16bf load_frag_b_pk(const uint32_t* s, int ld,
                                                       int k0, int n0) {
  int lane = (int)(threadIdx.x & 31);
  int n = n0 + (lane & 15);
  int kp0 = (k0 >> 1) + (lane >> 4) * 8;
  FragBF f;
#pragma unroll
  for (int j = 0; j < 8; ++j) f.u32[j] = s[(size_t)(kp0 + j) * ld + n];
  return f.v;
}

// B fragment 32x16 with B[k][n] = src[(n0+n)*ld + k0+k]  (transposed: rows are N)
static __device__ __forceinline__ v16bf load_frag_b_tr(const unsigned short* s, int ld,
                                                       int k0, int n0) {
  int lane = (int)(threadIdx.x & 31);
  int n = n0 + (lane & 15);
  int kh = (lane >> 4) * 16;
  FragBF f;
#pragma unroll
  for (int j = 0; j < 8; ++j)
    f.u32[j] = *(const uint32_t*)(s + (size_t)n * ld + k0 + kh + 2 * j);
  return f.v;
}

#if USE_TDM
// ---------- Tensor Data Mover: 2D tile global->LDS with row padding ----------
// D# per CDNA5 ISA ch.8. 6-arg builtin on this toolchain:
//   (uint32x4 g0, int32x8 g1, int32x4 g2, int32x4 g3, int32x8 g4, i32 cpol)
typedef __attribute__((ext_vector_type(4))) unsigned int v4u_;
typedef __attribute__((ext_vector_type(8))) int v8i_;
typedef __attribute__((ext_vector_type(4))) int v4i_;

static __device__ __forceinline__ void tdm_load_tile(
    uint32_t lds_addr, const void* gptr, int dsize_code /*0:1B 1:2B 2:4B*/,
    int dim0, int dim1, int stride0, int tile0, int tile1,
    int pad_interval_code, int pad_amount_code) {
  uint64_t ga = (uint64_t)(uintptr_t)gptr;
  v4u_ g0;
  g0[0] = 1u;                                  // count=1, user descriptor
  g0[1] = lds_addr;                            // LDS byte address
  g0[2] = (uint32_t)ga;                        // global_addr[31:0]
  g0[3] = (uint32_t)((ga >> 32) & 0x1FFFFFFu)  // global_addr[56:32]
        | (2u << 30);                          // type = 2 ("image")
  v8i_ g1;
  g1[0] = (dsize_code << 16) | (1 << 20)       // data_size, pad_enable
        | (pad_interval_code << 22) | (pad_amount_code << 25);
  g1[1] = (dim0 & 0xFFFF) << 16;                               // tensor_dim0[15:0]
  g1[2] = ((dim0 >> 16) & 0xFFFF) | ((dim1 & 0xFFFF) << 16);   // dim0 hi | dim1 lo
  g1[3] = ((dim1 >> 16) & 0xFFFF) | ((tile0 & 0xFFFF) << 16);  // dim1 hi | tile_dim0
  g1[4] = (tile1 & 0xFFFF);                                    // tile_dim1 (tile_dim2=0)
  g1[5] = stride0;                                             // tensor_dim0_stride lo
  g1[6] = 0;
  g1[7] = 0;
  v4i_ z4 = {0, 0, 0, 0};
  v8i_ z8 = {0, 0, 0, 0, 0, 0, 0, 0};
  __builtin_amdgcn_tensor_load_to_lds(g0, g1, z4, z4, z8, 0);
}
#endif

// =====================================================================
// Kernel 1: fused Q/K/V projection. grid = M/64; block = 256 (8 waves).
// Reads x ONCE and produces all three projections (6 WMMA tiles per wave).
// V output is row-pair-packed u32 for single-b32 B-fragment loads.
// =====================================================================
__global__ __launch_bounds__(256) void qkv_proj_kernel(
    const float* __restrict__ x, const float* __restrict__ Wq,
    const float* __restrict__ Wk, const float* __restrict__ Wv,
    unsigned short* __restrict__ qo, unsigned short* __restrict__ ko,
    uint32_t* __restrict__ vo) {
  const int row0 = blockIdx.x * 64;

  __shared__ __align__(16) unsigned short Xs[64][40];   // 64 rows x 32 K (+pad)
  __shared__ __align__(16) uint32_t Wp[3][16][68];      // per-proj K-pairs x N (+pad)

  const int tid = (int)threadIdx.x;
  const int lane = tid & 31, wave = tid >> 5;
  const int wr = (wave >> 1) * 16;
  const int wc = (wave & 1) * 32;

  v8f acc[3][2];
#pragma unroll
  for (int g = 0; g < 3; ++g)
#pragma unroll
    for (int t = 0; t < 2; ++t)
#pragma unroll
      for (int j = 0; j < 8; ++j) acc[g][t][j] = 0.f;

  for (int kk = 0; kk < NE; kk += 32) {
    for (int i = tid; i < 64 * 32; i += 256) {
      int r = i >> 5, c = i & 31;
      Xs[r][c] = f2bf(x[(size_t)(row0 + r) * NE + kk + c]);
    }
    for (int i = tid; i < 3 * 16 * 64; i += 256) {   // pack W row pairs (all 3)
      int g = i >> 10, kp = (i >> 6) & 15, c = i & 63;
      const float* W = (g == 0) ? Wq : (g == 1) ? Wk : Wv;
      float lo = W[(size_t)(kk + 2 * kp) * NH + c];
      float hi = W[(size_t)(kk + 2 * kp + 1) * NH + c];
      Wp[g][kp][c] = pack_bf2(lo, hi);
    }
    __syncthreads();

    v16bf a = load_frag_a(&Xs[0][0], 40, wr, 0);
#pragma unroll
    for (int g = 0; g < 3; ++g)
#pragma unroll
      for (int t = 0; t < 2; ++t) {
        v16bf bb = load_frag_b_pk(&Wp[g][0][0], 68, 0, wc + t * 16);
        acc[g][t] = bf16_wmma(a, bb, acc[g][t]);
      }
    __syncthreads();
  }

  // Q and K: plain bf16 row-major stores
#pragma unroll
  for (int t = 0; t < 2; ++t)
#pragma unroll
    for (int j = 0; j < 8; ++j) {
      int m = wr + (lane >> 4) * 8 + j;
      int n = wc + t * 16 + (lane & 15);
      qo[(size_t)(row0 + m) * NH + n] = f2bf(acc[0][t][j]);
      ko[(size_t)(row0 + m) * NH + n] = f2bf(acc[1][t][j]);
    }
  // V: row-pair-packed store; lane holds 8 consecutive rows so pairs are local
#pragma unroll
  for (int t = 0; t < 2; ++t)
#pragma unroll
    for (int j = 0; j < 8; j += 2) {
      int m = wr + (lane >> 4) * 8 + j;      // even
      int n = wc + t * 16 + (lane & 15);
      vo[(size_t)((row0 + m) >> 1) * NH + n] = pack_bf2(acc[2][t][j], acc[2][t][j + 1]);
    }
}

// =====================================================================
// Kernel 2: flash attention. grid = (T/64, B); block = 256 (8 waves).
// K/V tiles double-buffered via Tensor Data Mover: next block's DMA is
// issued before waiting on the current one (s_wait_tensorcnt 2), so the
// DMA overlaps WMMA + softmax. Row padding via D# pad fields.
// =====================================================================
__global__ __launch_bounds__(256) void flash_attn_kernel(
    const unsigned short* __restrict__ Q, const unsigned short* __restrict__ K,
    const uint32_t* __restrict__ V, const unsigned char* __restrict__ kpm,
    float* __restrict__ out) {
  const int b = (int)blockIdx.y;
  const int qt = (int)blockIdx.x;
  const int tid = (int)threadIdx.x;
  const int lane = tid & 31, wave = tid >> 5;
  const int wr = (wave >> 1) * 16;
  const int wc = (wave & 1) * 32;
  const size_t rowbase = (size_t)b * NT;
  const size_t vrowbase = (size_t)b * (NT / 2);

  __shared__ __align__(16) unsigned short Qs[64][72];
  __shared__ __align__(16) unsigned short Ks[2][64][72];   // double-buffered
  __shared__ __align__(16) uint32_t Vs[2][32][68];         // row-pair-packed, dbl-buf
  __shared__ __align__(16) unsigned short Ps[64][72];
  __shared__ float Ss[64][68];
  __shared__ float pmax[4][64], psum[4][64];
  __shared__ float mrow[64], lrow[64], arow[64];

#if USE_TDM
  if (wave == 0) {
    // Q tile: 64x64 bf16, rows contiguous; pad 32DW rows by 4DW -> ld 72 u16
    tdm_load_tile((uint32_t)(uintptr_t)&Qs[0][0], Q + (rowbase + qt * 64) * NH,
                  1, 64, 64, 64, 64, 64, /*32DW*/ 4, /*4DW*/ 3);
    tdm_load_tile((uint32_t)(uintptr_t)&Ks[0][0][0], K + rowbase * NH,
                  1, 64, 64, 64, 64, 64, 4, 3);
    // V tile: 32x64 u32; pad 64DW rows by 4DW -> ld 68 u32
    tdm_load_tile((uint32_t)(uintptr_t)&Vs[0][0][0], V + vrowbase * NH,
                  2, 64, 32, 64, 64, 32, /*64DW*/ 5, /*4DW*/ 3);
  }
#else
  for (int i = tid; i < 64 * 64; i += 256) {
    int r = i >> 6, c = i & 63;
    Qs[r][c] = Q[(rowbase + qt * 64 + r) * NH + c];
  }
#endif
  if (tid < 64) { mrow[tid] = -INFINITY; lrow[tid] = 0.f; }

  v8f acc[2];
#pragma unroll
  for (int t = 0; t < 2; ++t)
#pragma unroll
    for (int j = 0; j < 8; ++j) acc[t][j] = 0.f;

  for (int jb = 0; jb <= qt; ++jb) {           // causal early exit
    const int cur = jb & 1;
#if USE_TDM
    if (wave == 0) {
      if (jb < qt) {
        // prefetch next key block into the other buffer, then wait for current
        tdm_load_tile((uint32_t)(uintptr_t)&Ks[cur ^ 1][0][0],
                      K + (rowbase + (jb + 1) * 64) * NH, 1, 64, 64, 64, 64, 64, 4, 3);
        tdm_load_tile((uint32_t)(uintptr_t)&Vs[cur ^ 1][0][0],
                      V + (vrowbase + (jb + 1) * 32) * NH, 2, 64, 32, 64, 64, 32, 5, 3);
        __builtin_amdgcn_s_wait_tensorcnt(2);  // in-order: current K/V resident
      } else {
        __builtin_amdgcn_s_wait_tensorcnt(0);
      }
    }
#else
    for (int i = tid; i < 64 * 64; i += 256) {
      int r = i >> 6, c = i & 63;
      Ks[cur][r][c] = K[(rowbase + jb * 64 + r) * NH + c];
    }
    for (int i = tid; i < 32 * 64; i += 256) {
      int r = i >> 6, c = i & 63;
      Vs[cur][r][c] = V[(vrowbase + jb * 32 + r) * NH + c];
    }
#endif
    __syncthreads();

    // ---- S = Q K^T (each wave: 2 tiles of 16x16, K-dim 64 in two 32-chunks)
    v8f sc[2];
#pragma unroll
    for (int t = 0; t < 2; ++t)
#pragma unroll
      for (int j = 0; j < 8; ++j) sc[t][j] = 0.f;
#pragma unroll
    for (int kc = 0; kc < NH; kc += 32) {
      v16bf a = load_frag_a(&Qs[0][0], 72, wr, kc);
#pragma unroll
      for (int t = 0; t < 2; ++t) {
        v16bf bb = load_frag_b_tr(&Ks[cur][0][0], 72, kc, wc + t * 16);
        sc[t] = bf16_wmma(a, bb, sc[t]);
      }
    }
    // scale + causal & key-padding mask, spill to f32 LDS
#pragma unroll
    for (int t = 0; t < 2; ++t)
#pragma unroll
      for (int j = 0; j < 8; ++j) {
        int m = wr + (lane >> 4) * 8 + j;
        int n = wc + t * 16 + (lane & 15);
        int qg = qt * 64 + m, kg = jb * 64 + n;
        bool keep = (kg <= qg) && (kpm[(size_t)b * NT + kg] != 0);
        Ss[m][n] = keep ? sc[t][j] * SOFT_SCALE : -INFINITY;
      }
    __syncthreads();

    // ---- online softmax: pass1 row max (4 threads per row)
    {
      int row = tid & 63, sub = tid >> 6;
      float pm = -INFINITY;
#pragma unroll
      for (int c = 0; c < 16; ++c) pm = fmaxf(pm, Ss[row][sub * 16 + c]);
      pmax[sub][row] = pm;
    }
    __syncthreads();
    if (tid < 64) {
      float mo = mrow[tid];
      float mn = fmaxf(fmaxf(pmax[0][tid], pmax[1][tid]),
                       fmaxf(pmax[2][tid], pmax[3][tid]));
      mn = fmaxf(mo, mn);
      arow[tid] = (mn == -INFINITY) ? 0.f : __expf(mo - mn);
      mrow[tid] = mn;
    }
    __syncthreads();
    // pass2: P = exp(S - m_new) as packed bf16 pairs, partial row sums
    {
      int row = tid & 63, sub = tid >> 6;
      float mn = mrow[row], ps = 0.f;
#pragma unroll
      for (int c = 0; c < 16; c += 2) {
        int col = sub * 16 + c;
        float s0 = Ss[row][col], s1 = Ss[row][col + 1];
        float p0 = (s0 == -INFINITY) ? 0.f : __expf(s0 - mn);
        float p1 = (s1 == -INFINITY) ? 0.f : __expf(s1 - mn);
        *(uint32_t*)&Ps[row][col] = pack_bf2(p0, p1);
        ps += p0 + p1;
      }
      psum[sub][row] = ps;
    }
    __syncthreads();
    if (tid < 64)
      lrow[tid] = lrow[tid] * arow[tid] +
                  (psum[0][tid] + psum[1][tid] + psum[2][tid] + psum[3][tid]);

    // ---- rescale O, then O += P V
#pragma unroll
    for (int t = 0; t < 2; ++t)
#pragma unroll
      for (int j = 0; j < 8; ++j) {
        int m = wr + (lane >> 4) * 8 + j;
        acc[t][j] *= arow[m];
      }
#pragma unroll
    for (int kc = 0; kc < 64; kc += 32) {
      v16bf a = load_frag_a(&Ps[0][0], 72, wr, kc);
#pragma unroll
      for (int t = 0; t < 2; ++t) {
        v16bf bb = load_frag_b_pk(&Vs[cur][0][0], 68, kc, wc + t * 16);
        acc[t] = bf16_wmma(a, bb, acc[t]);
      }
    }
    __syncthreads();   // all reads of buf[cur] done before it is reused
  }

  // ---- O / l, store fp32
#pragma unroll
  for (int t = 0; t < 2; ++t)
#pragma unroll
    for (int j = 0; j < 8; ++j) {
      int m = wr + (lane >> 4) * 8 + j;
      int n = wc + t * 16 + (lane & 15);
      out[(rowbase + qt * 64 + m) * NH + n] = acc[t][j] / lrow[m];
    }
}

// =====================================================================
extern "C" void kernel_launch(void* const* d_in, const int* in_sizes, int n_in,
                              void* d_out, int out_size, void* d_ws, size_t ws_size,
                              hipStream_t stream) {
  (void)in_sizes; (void)n_in; (void)out_size; (void)ws_size;
  const float* x  = (const float*)d_in[0];
  const float* Wq = (const float*)d_in[1];
  const float* Wk = (const float*)d_in[2];
  const float* Wv = (const float*)d_in[3];
  const unsigned char* kpm = (const unsigned char*)d_in[4];
  float* out = (float*)d_out;

  unsigned short* qws = (unsigned short*)d_ws;                 // 2 MB bf16 Q
  unsigned short* kws = qws + (size_t)NB * NT * NH;            // 2 MB bf16 K
  uint32_t*       vws = (uint32_t*)(kws + (size_t)NB * NT * NH); // 2 MB packed V

  dim3 blk(256);
  dim3 g1((NB * NT) / 64);
  qkv_proj_kernel<<<g1, blk, 0, stream>>>(x, Wq, Wk, Wv, qws, kws, vws);

  dim3 g2(NT / 64, NB);
  flash_attn_kernel<<<g2, blk, 0, stream>>>(qws, kws, vws, kpm, out);
}
